// TwistorLNN_5660766896347
// MI455X (gfx1250) — matrix operand order repaired and checked by reading, not compiled, not run on hardware
//
#include <hip/hip_runtime.h>

#define T_STEPS 512
#define BATCH   256
#define INDIM   256
#define HID     512
#define ODIM    64

#define KCHUNK  128          // K-chunk staged in LDS per sync phase
#define LDSTR   144          // padded LDS row stride (bf16 elems): 288B -> 8-bank shift/row
#define LDSMAT  (64 * LDSTR) // per-matrix LDS extent (bf16 elems)

typedef __attribute__((ext_vector_type(16))) __bf16 v16bf;
typedef __attribute__((ext_vector_type(8)))  __bf16 v8bf;
typedef __attribute__((ext_vector_type(8)))  float  v8f;

__device__ __forceinline__ v8f wmma_bf16(v16bf a, v16bf b, v8f c) {
  return __builtin_amdgcn_wmma_f32_16x16x32_bf16(false, a, false, b, (short)0, c,
                                                 false, false);
}

// A operand (16x32 bf16). p = row_base + hi*16 (per-lane), k0 immediate.
__device__ __forceinline__ v16bf a_bf_ld(const __bf16* p, int k0) {
  v8bf a0 = *(const v8bf*)(p + k0);
  v8bf a1 = *(const v8bf*)(p + k0 + 8);
  return __builtin_shufflevector(a0, a1, 0,1,2,3,4,5,6,7,8,9,10,11,12,13,14,15);
}

// Same A layout but converting fp32 -> bf16 on the fly.
__device__ __forceinline__ v16bf a_f32_ld(const float* p, int k0) {
  v8f f0 = *(const v8f*)(p + k0);
  v8f f1 = *(const v8f*)(p + k0 + 8);
  v16bf r;
#pragma unroll
  for (int i = 0; i < 8; ++i) { r[i] = (__bf16)f0[i]; r[i + 8] = (__bf16)f1[i]; }
  return r;
}

// B operand (32x16 bf16) from global. p = base + n*ldb + hi*16.
__device__ __forceinline__ v16bf b_ld(const __bf16* p, int k0) {
  return *(const v16bf*)(p + k0);
}

__device__ __forceinline__ float sigmoidf_(float x) {
  return 1.0f / (1.0f + __expf(-x));
}

// ---------------- setup: weight prep + state init ----------------
__global__ void twistor_setup(const float* Wrw, const float* maskr,
                              const float* Wiw, const float* maski,
                              const float* Wtw, const float* Uw, const float* outw,
                              __bf16* Wr, __bf16* Wi, __bf16* Wt,
                              __bf16* Ubf, __bf16* outwbf,
                              float* zre, float* zim,
                              __bf16* tre, __bf16* tim, __bf16* zmod, __bf16* zrebf,
                              float* acc) {
  int i = blockIdx.x * blockDim.x + threadIdx.x;
  if (i < HID * HID) {
    Wr[i] = (__bf16)(Wrw[i] * sigmoidf_(maskr[i]));
    Wi[i] = (__bf16)(Wiw[i] * sigmoidf_(maski[i]));
    Wt[i] = (__bf16)Wtw[i];
  }
  if (i < HID * INDIM) Ubf[i] = (__bf16)Uw[i];
  if (i < ODIM * HID)  outwbf[i] = (__bf16)outw[i];
  if (i < BATCH * HID) {
    zre[i] = 0.0f; zim[i] = 0.0f;
    tre[i] = (__bf16)0.0f; tim[i] = (__bf16)0.0f;
    zmod[i] = (__bf16)1e-6f; zrebf[i] = (__bf16)0.0f;
  }
  if (i < 2) acc[i] = 0.0f;
}

// -------- off-critical-path: Ux = x @ U^T for ALL timesteps (fp32 out) -------
__global__ __launch_bounds__(256)
void twistor_ux_gemm(const float* __restrict__ x, const __bf16* __restrict__ Ubf,
                     float* __restrict__ Ux) {
  const int lane = threadIdx.x & 31;
  const int wave = threadIdx.x >> 5;
  const int l = lane & 15, hi = lane >> 4;
  const int m0 = (blockIdx.x * 8 + wave) * 16;   // row in [0, T*B)
  const int n0 = blockIdx.y * 64;

  v8f c[4] = {};
  const float* pA = x + (size_t)(m0 + l) * INDIM + hi * 16;
  const __bf16* pB[4];
#pragma unroll
  for (int s = 0; s < 4; ++s)
    pB[s] = Ubf + (size_t)(n0 + s * 16 + l) * INDIM + hi * 16;

#pragma unroll
  for (int k0 = 0; k0 < INDIM; k0 += 32) {
    v16bf a = a_f32_ld(pA, k0);
    v16bf b0 = b_ld(pB[0], k0), b1 = b_ld(pB[1], k0);
    v16bf b2 = b_ld(pB[2], k0), b3 = b_ld(pB[3], k0);
    c[0] = wmma_bf16(a, b0, c[0]);
    c[1] = wmma_bf16(a, b1, c[1]);
    c[2] = wmma_bf16(a, b2, c[2]);
    c[3] = wmma_bf16(a, b3, c[3]);
  }
#pragma unroll
  for (int s = 0; s < 4; ++s) {
    int n = n0 + s * 16 + l;
#pragma unroll
    for (int r = 0; r < 8; ++r)
      Ux[(size_t)(m0 + r + hi * 8) * HID + n] = c[s][r];
  }
}

// ---------------- per-step GEMMs + epilogue + norm partials ----------------
// 8 waves/block; block owns one 64-col N-strip. The 3 weight strips (shared by
// all 8 waves) are cooperatively staged in LDS per 128-K chunk; B operands
// then come from ds_load instead of per-WMMA L2 round trips.
__global__ __launch_bounds__(256)
void twistor_step_gemm(const float* __restrict__ x,
                       const float* __restrict__ UxPre,  // may be null
                       const float* __restrict__ zre, const float* __restrict__ zim,
                       const __bf16* __restrict__ tre, const __bf16* __restrict__ tim,
                       const __bf16* __restrict__ zmod,
                       const __bf16* __restrict__ Wr, const __bf16* __restrict__ Wi,
                       const __bf16* __restrict__ Wt, const __bf16* __restrict__ Ubf,
                       const float* __restrict__ Wrb, const float* __restrict__ Wib,
                       const float* __restrict__ Ub,  const float* __restrict__ Wtb,
                       const float* __restrict__ taub,
                       const float* __restrict__ breal, const float* __restrict__ bimag,
                       float* __restrict__ dre, float* __restrict__ dim_,
                       float* __restrict__ acc, int t) {
  __shared__ __bf16 ldsB[3 * LDSMAT];   // 3 * 64 * 144 * 2B = 54 KB

  const int lane = threadIdx.x & 31;
  const int wave = threadIdx.x >> 5;
  const int l = lane & 15, hi = lane >> 4;
  const int m0 = (blockIdx.x * 8 + wave) * 16;
  const int n0 = blockIdx.y * 64;

  // A-operand per-lane global base pointers.
  const __bf16* pAr = tre  + (size_t)(m0 + l) * HID + hi * 16;
  const __bf16* pAi = tim  + (size_t)(m0 + l) * HID + hi * 16;
  const __bf16* pAm = zmod + (size_t)(m0 + l) * HID + hi * 16;

  // B-operand per-lane LDS base pointers (rows are the N-strip's 64 columns).
  const __bf16 *qr[4], *qi[4], *qt[4];
#pragma unroll
  for (int s = 0; s < 4; ++s) {
    int row = s * 16 + l;
    qr[s] = &ldsB[0 * LDSMAT + row * LDSTR + hi * 16];
    qi[s] = &ldsB[1 * LDSMAT + row * LDSTR + hi * 16];
    qt[s] = &ldsB[2 * LDSMAT + row * LDSTR + hi * 16];
  }

  v8f cre[4] = {}, cim[4] = {}, ctau[4] = {}, cux[4] = {};

  if (!UxPre) {
    const float* pAx = x + ((size_t)t * BATCH + m0 + l) * INDIM + hi * 16;
    const __bf16* pBu[4];
#pragma unroll
    for (int s = 0; s < 4; ++s)
      pBu[s] = Ubf + (size_t)(n0 + s * 16 + l) * INDIM + hi * 16;
#pragma unroll
    for (int k0 = 0; k0 < INDIM; k0 += 32) {
      v16bf ax = a_f32_ld(pAx, k0);
      v16bf b0 = b_ld(pBu[0], k0), b1 = b_ld(pBu[1], k0);
      v16bf b2 = b_ld(pBu[2], k0), b3 = b_ld(pBu[3], k0);
      cux[0] = wmma_bf16(ax, b0, cux[0]);
      cux[1] = wmma_bf16(ax, b1, cux[1]);
      cux[2] = wmma_bf16(ax, b2, cux[2]);
      cux[3] = wmma_bf16(ax, b3, cux[3]);
    }
  }

  // 4 chunks of 128 K: cooperative global->LDS stage, then 4 WMMA K-iters.
  for (int chunk = 0; chunk < HID / KCHUNK; ++chunk) {
    const int kc = chunk * KCHUNK;
    // Stage 3 * 64 rows * 128 K bf16 = 3072 x 16B units; 12 units per thread.
#pragma unroll
    for (int j = 0; j < 12; ++j) {
      int u   = threadIdx.x + 256 * j;
      int mat = u >> 10;          // 1024 units per matrix (64 rows * 16 segs)
      int rem = u & 1023;
      int row = rem >> 4;
      int seg = rem & 15;
      const __bf16* base = (mat == 0) ? Wr : ((mat == 1) ? Wi : Wt);
      v8bf v = *(const v8bf*)(base + (size_t)(n0 + row) * HID + kc + seg * 8);
      *(v8bf*)&ldsB[mat * LDSMAT + row * LDSTR + seg * 8] = v;
    }
    __syncthreads();

#pragma unroll
    for (int kk = 0; kk < KCHUNK; kk += 32) {
      const int k0 = kc + kk;
      v16bf ar = a_bf_ld(pAr, k0);
      v16bf ai = a_bf_ld(pAi, k0);
      v16bf am = a_bf_ld(pAm, k0);
      v16bf br0 = *(const v16bf*)(qr[0] + kk), br1 = *(const v16bf*)(qr[1] + kk);
      v16bf br2 = *(const v16bf*)(qr[2] + kk), br3 = *(const v16bf*)(qr[3] + kk);
      v16bf bi0 = *(const v16bf*)(qi[0] + kk), bi1 = *(const v16bf*)(qi[1] + kk);
      v16bf bi2 = *(const v16bf*)(qi[2] + kk), bi3 = *(const v16bf*)(qi[3] + kk);
      v16bf bt0 = *(const v16bf*)(qt[0] + kk), bt1 = *(const v16bf*)(qt[1] + kk);
      v16bf bt2 = *(const v16bf*)(qt[2] + kk), bt3 = *(const v16bf*)(qt[3] + kk);
      cre[0]  = wmma_bf16(ar, br0, cre[0]);
      cre[1]  = wmma_bf16(ar, br1, cre[1]);
      cre[2]  = wmma_bf16(ar, br2, cre[2]);
      cre[3]  = wmma_bf16(ar, br3, cre[3]);
      cim[0]  = wmma_bf16(ai, bi0, cim[0]);
      cim[1]  = wmma_bf16(ai, bi1, cim[1]);
      cim[2]  = wmma_bf16(ai, bi2, cim[2]);
      cim[3]  = wmma_bf16(ai, bi3, cim[3]);
      ctau[0] = wmma_bf16(am, bt0, ctau[0]);
      ctau[1] = wmma_bf16(am, bt1, ctau[1]);
      ctau[2] = wmma_bf16(am, bt2, ctau[2]);
      ctau[3] = wmma_bf16(am, bt3, ctau[3]);
    }
    __syncthreads();  // protect LDS before next chunk's overwrite
  }

  // epilogue: dz, tau, clip, store d, accumulate norm
  const float* uxrow = UxPre ? (UxPre + ((size_t)t * BATCH) * HID) : nullptr;
  float lsum = 0.0f;
#pragma unroll
  for (int s = 0; s < 4; ++s) {
    int n = n0 + s * 16 + l;
    float ub = Ub[n], wrb = Wrb[n], wib = Wib[n], wtb = Wtb[n];
    float tb = taub[n], bre = breal[n], bim = bimag[n];
#pragma unroll
    for (int r = 0; r < 8; ++r) {
      int m = m0 + r + hi * 8;
      size_t idx = (size_t)m * HID + n;
      float ux  = (uxrow ? uxrow[idx] : cux[s][r]) + ub;
      float dzr = -zre[idx] + cre[s][r] + wrb + ux + bre;
      float dzi = -zim[idx] + cim[s][r] + wib + ux + bim;
      float tv  = sigmoidf_(ctau[s][r] + wtb) + tb;
      tv = fminf(fmaxf(tv, 0.01f), 1.0f) + 1e-6f;
      float drv = fminf(fmaxf(dzr / tv, -10.0f), 10.0f);
      float div = fminf(fmaxf(dzi / tv, -10.0f), 10.0f);
      dre[idx] = drv;
      dim_[idx] = div;
      lsum += sqrtf(drv * drv + div * div + 1e-12f);
    }
  }
#pragma unroll
  for (int off = 16; off > 0; off >>= 1) lsum += __shfl_down(lsum, off, 32);
  if (lane == 0) atomicAdd(&acc[t & 1], lsum);
}

// ---------------- per-step: scale finalize + Euler update ----------------
__global__ void twistor_step_update(const float* __restrict__ dre,
                                    const float* __restrict__ dim_,
                                    float* __restrict__ zre, float* __restrict__ zim,
                                    __bf16* __restrict__ tre, __bf16* __restrict__ tim,
                                    __bf16* __restrict__ zmod, __bf16* __restrict__ zrebf,
                                    float* __restrict__ acc, int t) {
  int i = blockIdx.x * blockDim.x + threadIdx.x;
  float mean = acc[t & 1] * (1.0f / (float)(BATCH * HID));
  float scale = (mean > 5.0f) ? (5.0f / (mean + 1e-6f)) : 1.0f;
  float r = zre[i] + 0.1f * (dre[i] * scale);
  float m = zim[i] + 0.1f * (dim_[i] * scale);
  r = fminf(fmaxf(r, -100.0f), 100.0f);
  m = fminf(fmaxf(m, -100.0f), 100.0f);
  zre[i] = r; zim[i] = m;
  tre[i]  = (__bf16)tanhf(r);
  tim[i]  = (__bf16)tanhf(m);
  zmod[i] = (__bf16)sqrtf(r * r + m * m + 1e-12f);
  zrebf[i] = (__bf16)r;
  if (i == 0) acc[(t + 1) & 1] = 0.0f;  // prep double-buffered accumulator
}

// ---------------- per-step: output projection y = z_re @ out_w^T + b ----------
__global__ __launch_bounds__(256)
void twistor_step_out(const __bf16* __restrict__ zrebf,
                      const __bf16* __restrict__ outwbf,
                      const float* __restrict__ outb,
                      float* __restrict__ y, int t) {
  const int lane = threadIdx.x & 31;
  const int wave = threadIdx.x >> 5;
  const int l = lane & 15, hi = lane >> 4;
  const int m0 = (blockIdx.x * 8 + wave) * 16;

  v8f c[4] = {};
  const __bf16* pA = zrebf + (size_t)(m0 + l) * HID + hi * 16;
  const __bf16* pB[4];
#pragma unroll
  for (int s = 0; s < 4; ++s)
    pB[s] = outwbf + (size_t)(s * 16 + l) * HID + hi * 16;

#pragma unroll 4
  for (int k0 = 0; k0 < HID; k0 += 32) {
    v16bf a = a_bf_ld(pA, k0);
    v16bf b0 = b_ld(pB[0], k0), b1 = b_ld(pB[1], k0);
    v16bf b2 = b_ld(pB[2], k0), b3 = b_ld(pB[3], k0);
    c[0] = wmma_bf16(a, b0, c[0]);
    c[1] = wmma_bf16(a, b1, c[1]);
    c[2] = wmma_bf16(a, b2, c[2]);
    c[3] = wmma_bf16(a, b3, c[3]);
  }
  float* yt = y + (size_t)t * BATCH * ODIM;
#pragma unroll
  for (int s = 0; s < 4; ++s) {
    int n = s * 16 + l;
    float ob = outb[n];
#pragma unroll
    for (int r = 0; r < 8; ++r) {
      int m = m0 + r + hi * 8;
      yt[(size_t)m * ODIM + n] = c[s][r] + ob;
    }
  }
}

extern "C" void kernel_launch(void* const* d_in, const int* in_sizes, int n_in,
                              void* d_out, int out_size, void* d_ws, size_t ws_size,
                              hipStream_t stream) {
  (void)in_sizes; (void)n_in; (void)out_size;
  const float* x     = (const float*)d_in[0];
  const float* Wrw   = (const float*)d_in[1];
  const float* Wrb   = (const float*)d_in[2];
  const float* Wiw   = (const float*)d_in[3];
  const float* Wib   = (const float*)d_in[4];
  const float* Uw    = (const float*)d_in[5];
  const float* Ub    = (const float*)d_in[6];
  const float* Wtw   = (const float*)d_in[7];
  const float* Wtb   = (const float*)d_in[8];
  const float* maskr = (const float*)d_in[9];
  const float* maski = (const float*)d_in[10];
  const float* taub  = (const float*)d_in[11];
  const float* breal = (const float*)d_in[12];
  const float* bimag = (const float*)d_in[13];
  const float* outw  = (const float*)d_in[14];
  const float* outb  = (const float*)d_in[15];
  float* y = (float*)d_out;

  char* w = (char*)d_ws;
  size_t off = 0;
  auto carve = [&](size_t bytes) -> char* {
    char* p = w + off;
    off += (bytes + 255) & ~(size_t)255;
    return p;
  };
  float*  acc   = (float*)carve(2 * sizeof(float));
  float*  zre   = (float*)carve((size_t)BATCH * HID * 4);
  float*  zim   = (float*)carve((size_t)BATCH * HID * 4);
  float*  dre   = (float*)carve((size_t)BATCH * HID * 4);
  float*  dim_  = (float*)carve((size_t)BATCH * HID * 4);
  __bf16* tre   = (__bf16*)carve((size_t)BATCH * HID * 2);
  __bf16* tim   = (__bf16*)carve((size_t)BATCH * HID * 2);
  __bf16* zmod  = (__bf16*)carve((size_t)BATCH * HID * 2);
  __bf16* zrebf = (__bf16*)carve((size_t)BATCH * HID * 2);
  __bf16* Wr    = (__bf16*)carve((size_t)HID * HID * 2);
  __bf16* Wi    = (__bf16*)carve((size_t)HID * HID * 2);
  __bf16* Wt    = (__bf16*)carve((size_t)HID * HID * 2);
  __bf16* Ubf   = (__bf16*)carve((size_t)HID * INDIM * 2);
  __bf16* outwbf= (__bf16*)carve((size_t)ODIM * HID * 2);

  // Optional big Ux buffer (takes x@U^T off the serial critical path).
  const size_t ux_bytes = (size_t)T_STEPS * BATCH * HID * sizeof(float);
  float* UxPre = nullptr;
  if (ws_size >= off + ux_bytes) UxPre = (float*)carve(ux_bytes);

  twistor_setup<<<(HID * HID) / 256, 256, 0, stream>>>(
      Wrw, maskr, Wiw, maski, Wtw, Uw, outw,
      Wr, Wi, Wt, Ubf, outwbf,
      zre, zim, tre, tim, zmod, zrebf, acc);

  if (UxPre) {
    twistor_ux_gemm<<<dim3((T_STEPS * BATCH) / 128, HID / 64), 256, 0, stream>>>(
        x, Ubf, UxPre);
  }

  for (int t = 0; t < T_STEPS; ++t) {
    twistor_step_gemm<<<dim3(BATCH / 128, HID / 64), 256, 0, stream>>>(
        x, UxPre, zre, zim, tre, tim, zmod, Wr, Wi, Wt, Ubf,
        Wrb, Wib, Ub, Wtb, taub, breal, bimag, dre, dim_, acc, t);
    twistor_step_update<<<(BATCH * HID) / 256, 256, 0, stream>>>(
        dre, dim_, zre, zim, tre, tim, zmod, zrebf, acc, t);
    twistor_step_out<<<BATCH / 128, 256, 0, stream>>>(zrebf, outwbf, outb, y, t);
  }
}
